// MemTransformerLM_16681652978134
// MI455X (gfx1250) — compile-verified, matrix operand order checked by reading, not run on hardware
//
#include <hip/hip_runtime.h>
#include <hip/hip_bf16.h>

// ---------------------------------------------------------------------------
// Transformer-XL dpfp-attention layer for gfx1250 (MI455X), wave32 + WMMA.
// Matmuls: v_wmma_f32_16x16x32_f16. GEMM tiles staged with
// global_load_async_to_lds_b128 (ASYNCcnt) + double-buffered LDS.
// ---------------------------------------------------------------------------

typedef __attribute__((ext_vector_type(16))) _Float16 v16h;
typedef __attribute__((ext_vector_type(8)))  float    v8f;

#define QL    512
#define ML    512
#define KL    1024
#define BN    8
#define DMD   1024
#define HN    16
#define DH    64
#define FDIM  384            // 2*DH*NROLL
#define SCALEF 0.125f        // 1/sqrt(64)
#define EPSF   1e-5f

// ---- CDNA5 async copy helpers ---------------------------------------------
// Per-lane 16B global->LDS DMA, tracked by ASYNCcnt (CDNA5 ISA).
// Generic pointers to __shared__ carry the LDS offset in their low 32 bits.
__device__ inline void async_g2l_b128(void* lds_ptr, const void* gptr) {
  const unsigned lds_off = (unsigned)(size_t)lds_ptr;
  asm volatile("global_load_async_to_lds_b128 %0, %1, off"
               :: "v"(lds_off), "v"(gptr) : "memory");
}
__device__ inline void wait_async0() {
  asm volatile("s_wait_asynccnt 0x0" ::: "memory");
}

// ---- WMMA fragment helpers (CDNA5 ISA 7.12.2 layouts) ----------------------
// A (16x32 f16): lane L -> row = L%16, kh = L/16; elem e -> k = (e>>3)*16 + kh*8 + (e&7)
// B (32x16 f16): mirrored (lane -> col); both load as "16 rows x K" row-major.
// C/D (16x16 f32): lane L -> col = L%16, vgpr r -> row = r + 8*(L/16).
__device__ inline v16h frag_rowK(const _Float16* base, int ld, int k0) {
  const int lane = threadIdx.x & 31;
  const int r    = lane & 15;
  const int kh   = lane >> 4;
  const _Float16* p = base + (size_t)r * ld + k0 + kh * 8;
  v16h f;
#pragma unroll
  for (int e = 0; e < 8; ++e) {    // two contiguous 16B chunks -> b128 loads
    f[e]     = p[e];
    f[e + 8] = p[e + 16];
  }
  return f;
}

// Same mapping but source rows are f32 (LDS tile staged by async DMA);
// convert to f16 during the fragment build (v_cvt_pk_f16_f32).
__device__ inline v16h frag_rowK_f32(const float* base, int ld, int k0) {
  const int lane = threadIdx.x & 31;
  const int r    = lane & 15;
  const int kh   = lane >> 4;
  const float* p = base + (size_t)r * ld + k0 + kh * 8;
  v16h f;
#pragma unroll
  for (int e = 0; e < 8; ++e) {
    f[e]     = (_Float16)p[e];
    f[e + 8] = (_Float16)p[e + 16];
  }
  return f;
}

__device__ inline v8f wmma_f16(const v16h& a, const v16h& b, const v8f& c) {
  return __builtin_amdgcn_wmma_f32_16x16x32_f16(false, a, false, b,
                                                (short)0, c, false, false);
}

// ---------------------------------------------------------------------------
// C[M,N] = X[M,K] @ W[N,K]^T (+ residual if RES), f32 in/out, f16 WMMA inner
// product. X rows come from A0 for row < rowsA0 else A1 (concat(mems,h)).
// Block: 256 threads = 8 waves (2 M x 4 N), tile 64x128, K step 32.
// Tiles staged as f32 via async global->LDS DMA, double buffered so the
// copies for step k+1 overlap the WMMAs of step k.
// ---------------------------------------------------------------------------
#define GT_M 64
#define GT_N 128
#define GT_K 32

__device__ inline void stage_tiles_async(
    float (*As)[GT_K], float (*Bs)[GT_K],
    const float* A0, const float* A1, int rowsA0,
    const float* W, int bm, int bn, int k0, int K)
{
  // A tile: 64x32 f32 = 512 x 16B chunks; 2 per thread
#pragma unroll
  for (int it = 0; it < 2; ++it) {
    const int idx = threadIdx.x + it * 256;
    const int row = idx >> 3;
    const int c4  = (idx & 7) << 2;
    const int grow = bm + row;
    const float* srow = (grow < rowsA0)
        ? A0 + (size_t)grow * K
        : A1 + (size_t)(grow - rowsA0) * K;
    async_g2l_b128(&As[row][c4], srow + k0 + c4);
  }
  // W tile: 128x32 f32 = 1024 chunks; 4 per thread
#pragma unroll
  for (int it = 0; it < 4; ++it) {
    const int idx = threadIdx.x + it * 256;
    const int row = idx >> 3;
    const int c4  = (idx & 7) << 2;
    async_g2l_b128(&Bs[row][c4], W + (size_t)(bn + row) * K + k0 + c4);
  }
}

template <int RES>
__global__ __launch_bounds__(256) void gemm_xwt_kernel(
    const float* __restrict__ A0, const float* __restrict__ A1, int rowsA0,
    const float* __restrict__ W, const float* __restrict__ residual,
    float* __restrict__ C, int M, int N, int K)
{
  __shared__ float As[2][GT_M][GT_K];   // 16 KB
  __shared__ float Bs[2][GT_N][GT_K];   // 32 KB

  const int bm   = blockIdx.y * GT_M;
  const int bn   = blockIdx.x * GT_N;
  const int wave = threadIdx.x >> 5;
  const int lane = threadIdx.x & 31;
  const int wm   = (wave >> 2) * 32;   // 0 / 32
  const int wn   = (wave & 3) * 32;    // 0..96

  v8f acc[2][2] = {};

  // prologue: stage k0 = 0 into buffer 0
  stage_tiles_async(As[0], Bs[0], A0, A1, rowsA0, W, bm, bn, 0, K);
  wait_async0();
  __syncthreads();

  int buf = 0;
  for (int k0 = 0; k0 < K; k0 += GT_K) {
    const int nbuf = buf ^ 1;
    if (k0 + GT_K < K)   // kick off next tile; overlaps with WMMAs below
      stage_tiles_async(As[nbuf], Bs[nbuf], A0, A1, rowsA0, W,
                        bm, bn, k0 + GT_K, K);

    const v16h af0 = frag_rowK_f32(&As[buf][wm][0],      GT_K, 0);
    const v16h af1 = frag_rowK_f32(&As[buf][wm + 16][0], GT_K, 0);
    const v16h bf0 = frag_rowK_f32(&Bs[buf][wn][0],      GT_K, 0);
    const v16h bf1 = frag_rowK_f32(&Bs[buf][wn + 16][0], GT_K, 0);
    acc[0][0] = wmma_f16(af0, bf0, acc[0][0]);
    acc[0][1] = wmma_f16(af0, bf1, acc[0][1]);
    acc[1][0] = wmma_f16(af1, bf0, acc[1][0]);
    acc[1][1] = wmma_f16(af1, bf1, acc[1][1]);

    wait_async0();        // this wave's DMA for nbuf is done
    __syncthreads();      // everyone's DMA is done; old buf fully consumed
    buf = nbuf;
  }

  const int nL = lane & 15, mh = lane >> 4;
#pragma unroll
  for (int im = 0; im < 2; ++im)
#pragma unroll
    for (int jn = 0; jn < 2; ++jn) {
      const int r0 = bm + wm + im * 16;
      const int c0 = bn + wn + jn * 16;
#pragma unroll
      for (int r = 0; r < 8; ++r) {
        const int row = r0 + r + 8 * mh;
        const int col = c0 + nL;
        float v = acc[im][jn][r];
        if (RES) v += residual[(size_t)row * N + col];
        C[(size_t)row * N + col] = v;
      }
    }
}

// ---------------------------------------------------------------------------
// dpfp feature map: x = [relu(q); relu(-q)] (128), out[f] = x[t]*x[(t-r)&127]
// for r=1..3. One block per (seq*batch row, head). Output f16 laid out
// [b*HN + h][seq][FDIM] so the attention GEMM reads contiguous K.
// ---------------------------------------------------------------------------
__global__ __launch_bounds__(128) void dpfp_kernel(
    const float* __restrict__ src, int ld, int col_off, int len,
    _Float16* __restrict__ out)
{
  __shared__ float x[128];
  const int row = blockIdx.x;        // i*BN + b
  const int hh  = blockIdx.y;
  const int t   = threadIdx.x;
  const int i   = row / BN;
  const int b   = row % BN;

  const float* s = src + (size_t)row * ld + col_off + hh * DH;
  if (t < DH) {
    const float v = s[t];
    x[t]      = fmaxf(v, 0.f);
    x[t + DH] = fmaxf(-v, 0.f);
  }
  __syncthreads();
  const float xt = x[t];
  _Float16* o = out + ((size_t)(b * HN + hh) * len + i) * FDIM;
  o[t]       = (_Float16)(xt * x[(t - 1) & 127]);
  o[t + 128] = (_Float16)(xt * x[(t - 2) & 127]);
  o[t + 256] = (_Float16)(xt * x[(t - 3) & 127]);
}

// ---------------------------------------------------------------------------
// Fused attention: per (b,h) head and 64-row i-tile.
//   S = qf @ kf^T * SCALE, masked (j <= i+MLEN); denom[i] += rowsum(S);
//   acc += S @ V ; final out = acc / (denom + eps).
// 8 waves: S phase -> wave (i_t = w&3, j-col tile = w>>2) over K=384 (12 WMMA);
// PV phase -> wave (i_t = w&3, d-half = w>>2), K=32 per j-tile (2 WMMA).
// S round-trips through LDS (f16) for the C->A layout change; V is staged
// transposed in LDS so B fragments are contiguous b128 loads.
// ---------------------------------------------------------------------------
__global__ __launch_bounds__(256) void attn_kernel(
    const _Float16* __restrict__ qf,   // [BN*HN][QL][FDIM]
    const _Float16* __restrict__ kf,   // [BN*HN][KL][FDIM]
    const float*    __restrict__ kv,   // [KL*BN][2*DMD]; v at col DMD + h*DH
    float* __restrict__ attn_vec)      // [QL*BN][DMD]
{
  __shared__ _Float16 Slds[64][32];
  __shared__ _Float16 Vt[64][32];      // Vt[d][j]
  __shared__ float denom[64];

  const int bh = blockIdx.x;
  const int b  = bh >> 4;              // / HN
  const int hh = bh & 15;              // % HN
  const int i0 = blockIdx.y * 64;
  const int wave = threadIdx.x >> 5;
  const int lane = threadIdx.x & 31;

  const int i_t = wave & 3;
  const int n_t = wave >> 2;           // j-column tile (S phase)
  const int d_t = wave >> 2;           // d half (PV phase)

  if (threadIdx.x < 64) denom[threadIdx.x] = 0.f;

  // preload qf A-fragments for this wave's i-tile (K = 384 -> 12 frags)
  v16h aq[FDIM / 32];
  const _Float16* qbase = qf + ((size_t)bh * QL + (i0 + i_t * 16)) * FDIM;
#pragma unroll
  for (int kk = 0; kk < FDIM / 32; ++kk)
    aq[kk] = frag_rowK(qbase, FDIM, kk * 32);

  v8f acc[2] = {};
  __syncthreads();

  const _Float16* kfh = kf + (size_t)bh * KL * FDIM;

  for (int j0 = 0; j0 < KL; j0 += 32) {
    // stage V transposed: Vt[d][j] = v[j0+j, b, hh, d]  (f32 -> f16)
    {
      const int j  = threadIdx.x >> 3;
      const int d0 = (threadIdx.x & 7) << 3;
      const float* vr = kv + ((size_t)(j0 + j) * BN + b) * (2 * DMD)
                           + DMD + hh * DH + d0;
#pragma unroll
      for (int e = 0; e < 8; ++e) Vt[d0 + e][j] = (_Float16)vr[e];
    }

    // prefetch next j-tile of kf (global_prefetch_b8)
    if (j0 + 32 < KL)
      __builtin_prefetch(kfh + (size_t)(j0 + 32 + n_t * 16 + (lane & 15)) * FDIM, 0, 1);

    // S tile: 16x16, K=384
    v8f sacc = {};
    const _Float16* kbase = kfh + (size_t)(j0 + n_t * 16) * FDIM;
#pragma unroll
    for (int kk = 0; kk < FDIM / 32; ++kk) {
      const v16h bk = frag_rowK(kbase, FDIM, kk * 32);
      sacc = wmma_f16(aq[kk], bk, sacc);
    }

    // mask (j <= i+MLEN), scale, spill to LDS f16
    const int jl = n_t * 16 + (lane & 15);
    const int jg = j0 + jl;
    const int mh = lane >> 4;
#pragma unroll
    for (int r = 0; r < 8; ++r) {
      const int il = i_t * 16 + r + 8 * mh;
      const int ig = i0 + il;
      const float sv = (jg <= ig + ML) ? sacc[r] * SCALEF : 0.f;
      Slds[il][jl] = (_Float16)sv;
    }
    __syncthreads();

    // denominator accumulation (same f16 values P@V consumes)
    if (threadIdx.x < 64) {
      float s = 0.f;
#pragma unroll
      for (int j = 0; j < 32; ++j) s += (float)Slds[threadIdx.x][j];
      denom[threadIdx.x] += s;
    }

    // P @ V for this j-tile (K = 32)
    const v16h as = frag_rowK(&Slds[i_t * 16][0], 32, 0);
#pragma unroll
    for (int tt = 0; tt < 2; ++tt) {
      const v16h bv = frag_rowK(&Vt[d_t * 32 + tt * 16][0], 32, 0);
      acc[tt] = wmma_f16(as, bv, acc[tt]);
    }
    __syncthreads();
  }

  // epilogue: divide by (denom + eps), write attn_vec[(i,b)][hh*64 + d]
  const int nL = lane & 15, mh = lane >> 4;
#pragma unroll
  for (int tt = 0; tt < 2; ++tt) {
    const int dcol = d_t * 32 + tt * 16 + nL;
#pragma unroll
    for (int r = 0; r < 8; ++r) {
      const int il = i_t * 16 + r + 8 * mh;
      const float dv = denom[il] + EPSF;
      attn_vec[((size_t)(i0 + il) * BN + b) * DMD + hh * DH + dcol] =
          acc[tt][r] / dv;
    }
  }
}

// ---------------------------------------------------------------------------
// Post-LN: per row of x[4096][1024] -> (x-mu)/sqrt(var+eps)*gamma + beta
// ---------------------------------------------------------------------------
__global__ __launch_bounds__(256) void ln_kernel(
    const float* __restrict__ x, const float* __restrict__ g,
    const float* __restrict__ bt, float* __restrict__ out)
{
  __shared__ float red[256];
  const int row = blockIdx.x;
  const int t   = threadIdx.x;
  const float* xr = x + (size_t)row * DMD;
  float vals[4];
  float s = 0.f;
#pragma unroll
  for (int c = 0; c < 4; ++c) { vals[c] = xr[t + c * 256]; s += vals[c]; }
  red[t] = s; __syncthreads();
  for (int st = 128; st > 0; st >>= 1) {
    if (t < st) red[t] += red[t + st];
    __syncthreads();
  }
  const float mu = red[0] * (1.f / DMD);
  __syncthreads();
  float vs = 0.f;
#pragma unroll
  for (int c = 0; c < 4; ++c) { const float d = vals[c] - mu; vs += d * d; }
  red[t] = vs; __syncthreads();
  for (int st = 128; st > 0; st >>= 1) {
    if (t < st) red[t] += red[t + st];
    __syncthreads();
  }
  const float inv = rsqrtf(red[0] * (1.f / DMD) + EPSF);
#pragma unroll
  for (int c = 0; c < 4; ++c) {
    const int col = t + c * 256;
    out[(size_t)row * DMD + col] = (vals[c] - mu) * inv * g[col] + bt[col];
  }
}

// ---------------------------------------------------------------------------
extern "C" void kernel_launch(void* const* d_in, const int* in_sizes, int n_in,
                              void* d_out, int out_size, void* d_ws, size_t ws_size,
                              hipStream_t stream) {
  (void)in_sizes; (void)n_in; (void)out_size; (void)ws_size;
  const float* h    = (const float*)d_in[0];   // [QL,BN,DMD]
  const float* mems = (const float*)d_in[1];   // [ML,BN,DMD]
  const float* Wq   = (const float*)d_in[2];   // [DMD,DMD]
  const float* Wkv  = (const float*)d_in[3];   // [2*DMD,DMD]
  const float* Wo   = (const float*)d_in[4];   // [DMD,DMD]
  const float* gam  = (const float*)d_in[5];
  const float* bet  = (const float*)d_in[6];
  // d_in[7] = attn_mask (bool) unused: mask is analytic (j <= i + MLEN)

  char* ws = (char*)d_ws;
  size_t off = 0;
  auto take = [&](size_t bytes) -> void* {
    void* p = (void*)(ws + off);
    off += (bytes + 255) & ~(size_t)255;
    return p;
  };
  float*    kv  = (float*)   take((size_t)(KL * BN) * (2 * DMD) * 4); //  64 MB
  float*    qx  = (float*)   take((size_t)(QL * BN) * DMD * 4);       //  16 MB (q, then x)
  _Float16* qfb = (_Float16*)take((size_t)(BN * HN) * QL * FDIM * 2); //  50 MB
  _Float16* kfb = (_Float16*)take((size_t)(BN * HN) * KL * FDIM * 2); // 101 MB
  float*    av  = (float*)   take((size_t)(QL * BN) * DMD * 4);       //  16 MB

  // q = h @ Wq^T
  gemm_xwt_kernel<0><<<dim3(DMD / GT_N, (QL * BN) / GT_M), 256, 0, stream>>>(
      h, h, QL * BN, Wq, nullptr, qx, QL * BN, DMD, DMD);
  // kv = concat(mems, h) @ Wkv^T
  gemm_xwt_kernel<0><<<dim3((2 * DMD) / GT_N, (KL * BN) / GT_M), 256, 0, stream>>>(
      mems, h, ML * BN, Wkv, nullptr, kv, KL * BN, 2 * DMD, DMD);
  // qf = dpfp(q), kf = dpfp(k)
  dpfp_kernel<<<dim3(QL * BN, HN), 128, 0, stream>>>(qx, DMD, 0, QL, qfb);
  dpfp_kernel<<<dim3(KL * BN, HN), 128, 0, stream>>>(kv, 2 * DMD, 0, KL, kfb);
  // fused masked-score + rowsum + P@V
  attn_kernel<<<dim3(BN * HN, QL / 64), 256, 0, stream>>>(qfb, kfb, kv, av);
  // x = attn_vec @ Wo^T + h   (reuse qx)
  gemm_xwt_kernel<1><<<dim3(DMD / GT_N, (QL * BN) / GT_M), 256, 0, stream>>>(
      av, av, QL * BN, Wo, h, qx, QL * BN, DMD, DMD);
  // post-LN -> out
  ln_kernel<<<QL * BN, 256, 0, stream>>>(qx, gam, bet, (float*)d_out);
}
